// QMixer_18416819765939
// MI455X (gfx1250) — matrix operand order repaired and checked by reading, not compiled, not run on hardware
//
#include <hip/hip_runtime.h>
#include <stdint.h>

// ---------------------------------------------------------------------------
// QMixer fused forward for gfx1250 (CDNA5, wave32, WMMA).
//   GEMM: [BT x 256] @ [256 x 704] in bf16 WMMA (f32 accum), fused epilogue.
//   Weight cols: [0,512)=w1_W, [512,576)=b1_W, [576,640)=wf_W, [640,704)=v1_W
//   N padded to 768 (48 tiles) so 8 waves x 6 tiles divides evenly.
//   M = 32 rows per workgroup: each B fragment feeds 2 WMMAs (2 M-subtiles),
//   halving L2 weight traffic per row vs M = 16.
// ---------------------------------------------------------------------------

#define STATE_DIM 256
#define NCOLS     704          // 512 + 64 + 64 + 64 (real columns)
#define NTILES_P  48           // padded tile count (768 cols)
#define KSTEPS    8            // STATE_DIM / 32
#define WAVES     8
#define TPB       (WAVES * 32) // 256
#define TPW       6            // tiles per wave
#define MROWS     32           // rows per workgroup
#define A_LD      264          // bf16 elements per LDS row (256 + 8 pad)
#define Z_LD      712          // f32 elements per LDS row (704 + 8 pad)

typedef __bf16 bf16x16 __attribute__((ext_vector_type(16)));
typedef float  f32x8   __attribute__((ext_vector_type(8)));

struct Frag32B { uint4 lo; uint4 hi; };   // 32 bytes == one v16bf fragment per lane

__device__ __forceinline__ uint16_t f32_to_bf16(float f) {
    uint32_t u = __builtin_bit_cast(uint32_t, f);
    uint32_t r = u + 0x7FFFu + ((u >> 16) & 1u);   // round-to-nearest-even
    return (uint16_t)(r >> 16);
}

// Pack two floats into one dword of bf16x2; prefer the HW packed convert.
__device__ __forceinline__ uint32_t pack_bf16x2(float lo, float hi) {
#if __has_builtin(__builtin_amdgcn_cvt_pk_bf16_f32)
    typedef __bf16 bf16x2 __attribute__((ext_vector_type(2)));
    bf16x2 p = __builtin_amdgcn_cvt_pk_bf16_f32(lo, hi);
    return __builtin_bit_cast(uint32_t, p);
#else
    return (uint32_t)f32_to_bf16(lo) | ((uint32_t)f32_to_bf16(hi) << 16);
#endif
}

// ---------------------------------------------------------------------------
// Pack kernel: bf16 B-fragments in WMMA lane layout.
// Fragment f = t*KSTEPS + kk; per-lane 16 bf16 (32B): element i holds
//   lane < 16 : W[kk*32 + i,       t*16 + lane     ]
//   lane >= 16: W[kk*32 + 16 + i,  t*16 + lane - 16]
// Columns >= 704 are zero padding.
// ---------------------------------------------------------------------------
__global__ void qmix_pack_weights(const float* __restrict__ w1_W,
                                  const float* __restrict__ b1_W,
                                  const float* __restrict__ wf_W,
                                  const float* __restrict__ v1_W,
                                  uint16_t* __restrict__ wsB) {
    int g = blockIdx.x * blockDim.x + threadIdx.x;
    const int total = NTILES_P * KSTEPS * 32 * 16;
    if (g >= total) return;
    int i    =  g        & 15;
    int lane = (g >>  4) & 31;
    int kk   = (g >>  9) & 7;
    int t    =  g >> 12;

    int K = kk * 32 + ((lane < 16) ? i : (16 + i));
    int N = t * 16 + (lane & 15);

    float v = 0.f;
    if (N < 512)      v = w1_W[(size_t)K * 512 + N];
    else if (N < 576) v = b1_W[(size_t)K * 64 + (N - 512)];
    else if (N < 640) v = wf_W[(size_t)K * 64 + (N - 576)];
    else if (N < 704) v = v1_W[(size_t)K * 64 + (N - 640)];

    wsB[g] = f32_to_bf16(v);
}

__device__ __forceinline__ bf16x16 load_bfrag(const uint16_t* __restrict__ wsB,
                                              int t, int kk, int lane) {
    const uint4* p = reinterpret_cast<const uint4*>(
        wsB + (((size_t)t * KSTEPS + kk) * 32 + lane) * 16);
    Frag32B f;
    f.lo = p[0];
    f.hi = p[1];
    return __builtin_bit_cast(bf16x16, f);
}

// ---------------------------------------------------------------------------
// Main fused kernel: one block = 32 rows, 8 waves, 6 N-tiles x 2 M-subtiles
// per wave. B fragments software-pipelined one iteration deep.
// ---------------------------------------------------------------------------
__global__ void __launch_bounds__(TPB)
qmix_fused(const float* __restrict__ agent_qs,   // [BT,8]
           const float* __restrict__ states,     // [BT,256]
           const uint16_t* __restrict__ wsB,     // packed bf16 B fragments
           const float* __restrict__ w1_b,       // [512]
           const float* __restrict__ b1_b,       // [64]
           const float* __restrict__ wf_b,       // [64]
           const float* __restrict__ v1_b,       // [64]
           const float* __restrict__ v2_W,       // [64]
           const float* __restrict__ v2_b,       // [1]
           float* __restrict__ out,              // [BT]
           int BT) {
    // Aliased LDS: states tile (bf16, 16.5 KB) lives only until the K-loop
    // ends; the Z spill tile (f32, 89 KB) reuses the same region afterwards.
    __shared__ __align__(16) char smem[MROWS * Z_LD * 4];
    uint16_t* sS = reinterpret_cast<uint16_t*>(smem);   // [MROWS][A_LD] bf16
    float*    sZ = reinterpret_cast<float*>(smem);      // [MROWS][Z_LD] f32

    const int tid  = threadIdx.x;
    const int lane = tid & 31;
    const int wave = tid >> 5;
    const int bt0  = blockIdx.x * MROWS;

    // ---- Stage states tile -> LDS as bf16 (32 rows x 256 K) ----------------
    #pragma unroll
    for (int j = 0; j < 8; ++j) {
        int linear = j * TPB + tid;          // float4 index within the tile
        int row = linear >> 6;               // (linear*4) / 256
        int k   = (linear & 63) << 2;        // (linear*4) % 256
        float4 v4 = make_float4(0.f, 0.f, 0.f, 0.f);
        if (bt0 + row < BT) {
            v4 = *reinterpret_cast<const float4*>(states + (size_t)(bt0 + row) * STATE_DIM + k);
        }
        uint2 packed;
        packed.x = pack_bf16x2(v4.x, v4.y);
        packed.y = pack_bf16x2(v4.z, v4.w);
        *reinterpret_cast<uint2*>(&sS[row * A_LD + k]) = packed;
    }
    __syncthreads();

    // ---- WMMA K-loop -------------------------------------------------------
    f32x8 acc0[TPW];   // M-subtile 0 (rows 0-15)
    f32x8 acc1[TPW];   // M-subtile 1 (rows 16-31)
    #pragma unroll
    for (int j = 0; j < TPW; ++j) { acc0[j] = (f32x8)(0.0f); acc1[j] = (f32x8)(0.0f); }

    const int mrow  = lane & 15;
    const int tbase = wave * TPW;

    bf16x16 curB = load_bfrag(wsB, tbase, 0, lane);     // prime the pipeline

    #pragma unroll
    for (int kk = 0; kk < KSTEPS; ++kk) {
        // A fragments: two contiguous 8-element bf16 runs per lane.
        int kb0 = kk * 32 + ((lane >> 4) << 4);         // +0 / +16 per half-wave
        Frag32B fa0, fa1;
        fa0.lo = *reinterpret_cast<const uint4*>(&sS[mrow * A_LD + kb0]);
        fa0.hi = *reinterpret_cast<const uint4*>(&sS[mrow * A_LD + kb0 + 8]);
        fa1.lo = *reinterpret_cast<const uint4*>(&sS[(16 + mrow) * A_LD + kb0]);
        fa1.hi = *reinterpret_cast<const uint4*>(&sS[(16 + mrow) * A_LD + kb0 + 8]);
        bf16x16 a0 = __builtin_bit_cast(bf16x16, fa0);
        bf16x16 a1 = __builtin_bit_cast(bf16x16, fa1);

        #pragma unroll
        for (int j = 0; j < TPW; ++j) {
            // Issue next B load before consuming the current one (1-deep SWP).
            int jn = (j == TPW - 1) ? 0 : j + 1;
            int kn = (j == TPW - 1) ? ((kk == KSTEPS - 1) ? kk : kk + 1) : kk;
            bf16x16 nxtB = load_bfrag(wsB, tbase + jn, kn, lane);

            // Warm near caches two K-steps ahead (WGP-scope global_prefetch).
            if (kk < KSTEPS - 2) {
                __builtin_prefetch(
                    wsB + (((size_t)(tbase + j) * KSTEPS + kk + 2) * 32 + lane) * 16, 0, 3);
            }

            acc0[j] = __builtin_amdgcn_wmma_f32_16x16x32_bf16(
                false, a0, false, curB, (short)0, acc0[j], false, false);
            acc1[j] = __builtin_amdgcn_wmma_f32_16x16x32_bf16(
                false, a1, false, curB, (short)0, acc1[j], false, false);
            curB = nxtB;
        }
    }

    // states tile is dead; Z spill reuses the same LDS bytes.
    __syncthreads();

    // ---- Spill accumulators to LDS: Z[32][704] -----------------------------
    #pragma unroll
    for (int j = 0; j < TPW; ++j) {
        int t = tbase + j;
        if (t < NCOLS / 16) {                       // skip zero-padding tiles
            int col   = t * 16 + (lane & 15);
            int rbase = (lane < 16) ? 0 : 8;
            #pragma unroll
            for (int r = 0; r < 8; ++r) {
                sZ[(rbase + r) * Z_LD + col]      = acc0[j][r];
                sZ[(16 + rbase + r) * Z_LD + col] = acc1[j][r];
            }
        }
    }
    __syncthreads();

    // ---- Epilogue: 8 lanes per row, each handles 8 embed slots -------------
    const int row = tid >> 3;        // 0..31
    const int sub = tid & 7;         // 0..7
    const int bt  = bt0 + row;

    float q[8];
    if (bt < BT) {
        float4 q0 = *reinterpret_cast<const float4*>(agent_qs + (size_t)bt * 8);
        float4 q1 = *reinterpret_cast<const float4*>(agent_qs + (size_t)bt * 8 + 4);
        q[0] = q0.x; q[1] = q0.y; q[2] = q0.z; q[3] = q0.w;
        q[4] = q1.x; q[5] = q1.y; q[6] = q1.z; q[7] = q1.w;
    } else {
        #pragma unroll
        for (int a = 0; a < 8; ++a) q[a] = 0.f;
    }

    const float* zrow = &sZ[row * Z_LD];
    float acc_y = 0.f;
    #pragma unroll
    for (int e8 = 0; e8 < 8; ++e8) {
        int e = e8 * 8 + sub;
        float hsum = 0.f;
        #pragma unroll
        for (int a = 0; a < 8; ++a) {
            hsum += q[a] * fabsf(zrow[a * 64 + e] + w1_b[a * 64 + e]);
        }
        float b1v = zrow[512 + e] + b1_b[e];
        float h = hsum + b1v;
        h = (h > 0.f) ? h : (__expf(h) - 1.f);            // elu, alpha = 1
        float wfv = fabsf(zrow[576 + e] + wf_b[e]);
        float v1v = zrow[640 + e] + v1_b[e];
        v1v = (v1v > 0.f) ? v1v : 0.f;                    // relu
        acc_y += h * wfv + v1v * v2_W[e];
    }

    // width-8 reduction within the wave
    #pragma unroll
    for (int o = 1; o < 8; o <<= 1) {
        acc_y += __shfl_xor(acc_y, o, 8);
    }

    if (sub == 0 && bt < BT) {
        out[bt] = acc_y + v2_b[0];
    }
}

// ---------------------------------------------------------------------------
extern "C" void kernel_launch(void* const* d_in, const int* in_sizes, int n_in,
                              void* d_out, int out_size, void* d_ws, size_t ws_size,
                              hipStream_t stream) {
    const float* agent_qs = (const float*)d_in[0];
    const float* states   = (const float*)d_in[1];
    const float* w1_W     = (const float*)d_in[2];
    const float* w1_b     = (const float*)d_in[3];
    const float* b1_W     = (const float*)d_in[4];
    const float* b1_b     = (const float*)d_in[5];
    const float* wf_W     = (const float*)d_in[6];
    const float* wf_b     = (const float*)d_in[7];
    const float* v1_W     = (const float*)d_in[8];
    const float* v1_b     = (const float*)d_in[9];
    const float* v2_W     = (const float*)d_in[10];
    const float* v2_b     = (const float*)d_in[11];
    float* out = (float*)d_out;

    const int BT = in_sizes[0] / 8;                 // agent_qs = [BT, 8]
    uint16_t* wsB = (uint16_t*)d_ws;                // 48*8*32*16 bf16 = 384 KB

    const int packTotal = NTILES_P * KSTEPS * 32 * 16;
    qmix_pack_weights<<<(packTotal + 255) / 256, 256, 0, stream>>>(
        w1_W, b1_W, wf_W, v1_W, wsB);

    const int nBlocks = (BT + MROWS - 1) / MROWS;
    qmix_fused<<<nBlocks, TPB, 0, stream>>>(
        agent_qs, states, wsB, w1_b, b1_b, wf_b, v1_b, v2_W, v2_b, out, BT);
}
